// DeformConv_27178553049517
// MI455X (gfx1250) — matrix operand order repaired and checked by reading, not compiled, not run on hardware
//
#include <hip/hip_runtime.h>
#include <hip/hip_bf16.h>

typedef _Float16 v16h __attribute__((ext_vector_type(16)));
typedef _Float16 v8h  __attribute__((ext_vector_type(8)));
typedef float    v8f  __attribute__((ext_vector_type(8)));
typedef float    f2   __attribute__((ext_vector_type(2)));

#define CIN   256
#define COUT  256
#define HW    4096     // 64*64
#define KTOT  2304     // Cin * 9
#define KT32  72       // KTOT / 32
#define KPAD  2312     // padded LDS K stride (halves) to break bank alignment
#define NT    32       // spatial positions per workgroup

// ---------------------------------------------------------------------------
// Pack weight f32 (Cout, Cin, 3, 3) into the CDNA5 WMMA 16-bit A-matrix 16x32
// per-lane layout. GEMM K index is permuted to kk = kpos*256 + ci (tap-major)
// so the sampler can write 8 consecutive channels with one ds_store_b128;
// we apply the same permutation here so the contraction is unchanged.
// dst index = ((mtile*72 + ktile)*32 + lane)*16 + h, where element h of the
// lane's v16h maps to K = (h<8 ? h : h+8) + (lane>=16 ? 8 : 0).
// ---------------------------------------------------------------------------
__global__ __launch_bounds__(256) void pack_weight_f16(
    const float* __restrict__ w, _Float16* __restrict__ wp) {
  int idx  = blockIdx.x * 256 + threadIdx.x;      // 0 .. 16*72*32*16-1
  int h    = idx & 15;
  int lane = (idx >> 4) & 31;
  int kt   = (idx >> 9) % KT32;
  int mt   = (idx >> 9) / KT32;
  int m    = mt * 16 + (lane & 15);
  int koff = (h < 8 ? h : h + 8) + ((lane & 16) ? 8 : 0);
  int kk   = kt * 32 + koff;                      // < 2304 always
  int ci   = kk & 255;                            // channel
  int kpos = kk >> 8;                             // tap 0..8
  wp[idx]  = (_Float16)w[(size_t)m * KTOT + ci * 9 + kpos];
}

// Load a B fragment (16-bit B 32x16 layout) from LDS: lane's 16 halves are
// two 8-half chunks at K offsets 0 and 16; do it as two 16B-aligned chunks.
static __device__ __forceinline__ v16h lds_b_frag(const _Float16* p) {
  v8h lo = *(const v8h*)p;
  v8h hi = *(const v8h*)(p + 16);
  return __builtin_shufflevector(lo, hi, 0,1,2,3,4,5,6,7,8,9,10,11,12,13,14,15);
}

// ---------------------------------------------------------------------------
// Fused deformable-conv: build f16 cols tile (K=2304 x N=32) in LDS via
// bilinear sampling (2 x b64 gathers per channel-tap, corner selection folded
// into per-tap coefficients), then 8 waves x (2 Mtiles x 2 Ntiles) WMMA
// contraction over all Cout=256 rows.
// ---------------------------------------------------------------------------
__global__ __launch_bounds__(256) void deform_conv_wmma(
    const float* __restrict__ x, const float* __restrict__ offset,
    const _Float16* __restrict__ wp, float* __restrict__ out) {
  extern __shared__ __align__(16) _Float16 cols[];   // [NT][KPAD] halves

  const int tid  = threadIdx.x;
  const int s0   = blockIdx.x * NT;       // 32 consecutive spatial positions
  const int nloc = tid & 31;              // spatial position within tile
  const int cgrp = tid >> 5;              // channel group (32 ch per wave)

  // ---- decode spatial position (all 32 share b and yo: 64-wide rows) ----
  const int s  = s0 + nloc;
  const int b  = s >> 12;
  const int yo = (s >> 6) & 63;
  const int xo = s & 63;
  const int sp = yo * 64 + xo;

  const float* __restrict__ xb = x + (size_t)b * CIN * HW;
  const float* __restrict__ ob = offset + (size_t)b * 18 * HW;

  // ---- im2col with bilinear sampling into LDS (K layout: kpos*256 + ci) ----
  for (int k = 0; k < 9; ++k) {
    const int kh = k / 3, kw = k % 3;
    const float offy = ob[(size_t)(2 * k + 0) * HW + sp];
    const float offx = ob[(size_t)(2 * k + 1) * HW + sp];
    const float py = (float)(yo - 1 + kh) + offy;
    const float px = (float)(xo - 1 + kw) + offx;
    const float fy = floorf(py), fx = floorf(px);
    const float ly = py - fy,    lx = px - fx;
    const int y0 = (int)fy, x0i = (int)fx;
    const int y1 = y0 + 1,  x1i = x0i + 1;
    const float vy0 = (y0  >= 0 && y0  <= 63) ? 1.0f : 0.0f;
    const float vy1 = (y1  >= 0 && y1  <= 63) ? 1.0f : 0.0f;
    const float vx0 = (x0i >= 0 && x0i <= 63) ? 1.0f : 0.0f;
    const float vx1 = (x1i >= 0 && x1i <= 63) ? 1.0f : 0.0f;
    const int yi0 = min(max(y0, 0), 63),  yi1 = min(max(y1, 0), 63);
    const int xi0 = min(max(x0i, 0), 63), xi1 = min(max(x1i, 0), 63);
    const float w00 = (1.0f - ly) * (1.0f - lx) * vy0 * vx0;
    const float w01 = (1.0f - ly) * lx          * vy0 * vx1;
    const float w10 = ly          * (1.0f - lx) * vy1 * vx0;
    const float w11 = ly          * lx          * vy1 * vx1;

    // Fold corner selection into pair coefficients (per-tap, channel-uniform):
    // pair q = {p[ib], p[ib+1]} with ib = min(i, HW-2); clamped/invalid
    // corners always end up with a zero coefficient.
    const int d   = xi1 - xi0;                    // 0 or 1
    const int i00 = yi0 * 64 + xi0;
    const int i10 = yi1 * 64 + xi0;
    const int ib0 = min(i00, HW - 2);
    const int ib1 = min(i10, HW - 2);
    const int o0  = i00 - ib0;                    // 0 or 1
    const int o1  = i10 - ib1;                    // 0 or 1
    const float a0 = (o0     == 0 ? w00 : 0.0f) + (o0 + d == 0 ? w01 : 0.0f);
    const float b0 = (o0     == 1 ? w00 : 0.0f) + (o0 + d == 1 ? w01 : 0.0f);
    const float a1 = (o1     == 0 ? w10 : 0.0f) + (o1 + d == 0 ? w11 : 0.0f);
    const float b1 = (o1     == 1 ? w10 : 0.0f) + (o1 + d == 1 ? w11 : 0.0f);

    const float* __restrict__ pc = xb + (size_t)cgrp * 32 * HW;
    _Float16* __restrict__ dst = cols + (size_t)nloc * KPAD + k * 256 + cgrp * 32;
#pragma unroll
    for (int c8 = 0; c8 < 32; c8 += 8) {
      v8h packv;
#pragma unroll
      for (int c = 0; c < 8; ++c) {
        const float* p = pc + (size_t)(c8 + c) * HW;
        const f2 q0 = *(const f2*)(p + ib0);   // unaligned b64 gather (row y0)
        const f2 q1 = *(const f2*)(p + ib1);   // unaligned b64 gather (row y1)
        packv[c] = (_Float16)(a0 * q0[0] + b0 * q0[1] + a1 * q1[0] + b1 * q1[1]);
      }
      *(v8h*)(dst + c8) = packv;               // one 16B LDS store per 8 ch
    }
  }
  __syncthreads();

  // ---- WMMA GEMM: wave wv handles mtiles {2wv, 2wv+1} x ntiles {0,1} ----
  const int lane = tid & 31;
  const int wv   = tid >> 5;
  const int m0   = wv * 2;
  const int lrow = lane & 15;
  const int kshift = (lane & 16) ? 8 : 0;    // B layout: upper half-wave = K+8

  const _Float16* __restrict__ aw =
      wp + (size_t)m0 * KT32 * 512 + (size_t)lane * 16;  // A frag base (mtile m0)
  const _Float16* __restrict__ bp0 = cols + (size_t)(0 * 16 + lrow) * KPAD + kshift;
  const _Float16* __restrict__ bp1 = cols + (size_t)(1 * 16 + lrow) * KPAD + kshift;

  v8f acc00 = {}, acc01 = {}, acc10 = {}, acc11 = {};

#pragma unroll 2
  for (int kt = 0; kt < KT32; ++kt) {
    const v16h a0 = *(const v16h*)(aw + (size_t)kt * 512);
    const v16h a1 = *(const v16h*)(aw + (size_t)(KT32 + kt) * 512);
    const v16h b0 = lds_b_frag(bp0 + kt * 32);
    const v16h b1 = lds_b_frag(bp1 + kt * 32);
    acc00 = __builtin_amdgcn_wmma_f32_16x16x32_f16(false, a0, false, b0, (short)0, acc00, false, false);
    acc01 = __builtin_amdgcn_wmma_f32_16x16x32_f16(false, a0, false, b1, (short)0, acc01, false, false);
    acc10 = __builtin_amdgcn_wmma_f32_16x16x32_f16(false, a1, false, b0, (short)0, acc10, false, false);
    acc11 = __builtin_amdgcn_wmma_f32_16x16x32_f16(false, a1, false, b1, (short)0, acc11, false, false);
  }

  // ---- store D fragments: element (vgpr r, lane l) -> M = r + 8*(l>=16),
  //      N = l & 15.  Rows of 16 consecutive floats -> coalesced stores. ----
  const int nf    = lane & 15;
  const int mhalf = (lane & 16) ? 8 : 0;
  const int xbase = s0 & 63;
  float* __restrict__ orow =
      out + ((size_t)b * COUT) * HW + (size_t)yo * 64 + xbase + nf;

#pragma unroll
  for (int r = 0; r < 8; ++r) {
    const int m = r + mhalf;
    orow[(size_t)((m0 + 0) * 16 + m) * HW + 0]  = acc00[r];
    orow[(size_t)((m0 + 0) * 16 + m) * HW + 16] = acc01[r];
    orow[(size_t)((m0 + 1) * 16 + m) * HW + 0]  = acc10[r];
    orow[(size_t)((m0 + 1) * 16 + m) * HW + 16] = acc11[r];
  }
}

extern "C" void kernel_launch(void* const* d_in, const int* in_sizes, int n_in,
                              void* d_out, int out_size, void* d_ws, size_t ws_size,
                              hipStream_t stream) {
  const float* x      = (const float*)d_in[0];
  const float* offset = (const float*)d_in[1];
  const float* weight = (const float*)d_in[2];
  float*       out    = (float*)d_out;
  _Float16*    wp     = (_Float16*)d_ws;   // 16*72*32*16 halves = 1.125 MB

  // 1) pack weight into WMMA A-fragment layout (f32 -> f16, K permuted)
  pack_weight_f16<<<(16 * KT32 * 32 * 16) / 256, 256, 0, stream>>>(weight, wp);

  // 2) fused sample + WMMA GEMM; dynamic LDS = 32 * 2312 halves = 147968 B
  const int nblocks = (8 * HW) / NT;       // 32768 / 32 = 1024
  deform_conv_wmma<<<nblocks, 256, NT * KPAD * sizeof(_Float16), stream>>>(
      x, offset, wp, out);
}
